// Attention_Layer_64364379898508
// MI455X (gfx1250) — matrix operand order repaired and checked by reading, not compiled, not run on hardware
//
#include <hip/hip_runtime.h>
#include <hip/hip_bf16.h>

typedef __attribute__((ext_vector_type(16))) _Float16 v16h;
typedef __attribute__((ext_vector_type(8)))  _Float16 v8h;
typedef __attribute__((ext_vector_type(8)))  float    v8f;

#define B_   4096
#define T_   200
#define D_   64
#define H1_  80
#define H2_  40
#define TPAD 208

#define KSTRIDE  72    // halves: 64 + 8 pad (144B rows, 16B aligned, conflict-free)
#define W1STRIDE 264   // halves: 256 + 8 pad
#define HSTRIDE  104   // halves: 96 + 8 pad
#define W2STRIDE 104   // halves: 96 + 8 pad

#define NT1 5   // 80 / 16 output tiles, GEMM1
#define NK1 8   // 256 / 32 K-steps, GEMM1
#define NT2 3   // 48 / 16 output tiles, GEMM2 (40 padded to 48)
#define NK2 3   // 96 / 32 K-steps, GEMM2 (80 padded to 96)

__device__ __forceinline__ v8f wmma_f16(v16h a, v16h b, v8f c) {
  return __builtin_amdgcn_wmma_f32_16x16x32_f16(
      /*neg_a=*/false, a, /*neg_b=*/false, b,
      /*c_mod=*/(short)0, c, /*reuse_a=*/false, /*reuse_b=*/false);
}

__global__ __launch_bounds__(256) void attn_mlp_kernel(
    const float* __restrict__ q,  const float* __restrict__ k,
    const float* __restrict__ v,  const int*   __restrict__ mask,
    const float* __restrict__ W1, const float* __restrict__ b1,
    const float* __restrict__ W2, const float* __restrict__ b2,
    const float* __restrict__ Wf, const float* __restrict__ bf,
    float* __restrict__ out)
{
  __shared__ __align__(16) _Float16 s_k[TPAD * KSTRIDE];     // k[b] as f16
  __shared__ __align__(16) _Float16 s_q[64];
  __shared__ __align__(16) _Float16 s_w1t[H1_ * W1STRIDE];   // W1^T [80][256]
  __shared__ __align__(16) _Float16 s_w2t[48 * W2STRIDE];    // W2^T [48][96] zero-padded
  __shared__ __align__(16) _Float16 s_h[TPAD * HSTRIDE];     // hidden1 [208][96] zero-padded
  __shared__ float s_b1[H1_];
  __shared__ float s_b2[48];
  __shared__ float s_wf[48];
  __shared__ float s_bf;
  __shared__ float s_logits[TPAD];
  __shared__ float s_red[256];
  __shared__ float s_vpart[4][64];

  const int b    = blockIdx.x;
  const int tid  = threadIdx.x;
  const int lane = tid & 31;
  const int wv   = tid >> 5;

  const float* vb = v + (size_t)b * T_ * D_;

  // Prefetch v[b] (51.2 KB) so the final attn.v pass hits cache; overlaps MLP compute.
  for (int i = tid; i < (T_ * D_) / 32; i += 256)
    __builtin_prefetch((const void*)(vb + i * 32), 0, 0);

  // ---------------- Stage inputs into LDS ----------------
  if (tid < 64) s_q[tid] = (_Float16)q[(size_t)b * D_ + tid];

  for (int idx = tid; idx < T_ * D_; idx += 256) {
    int t = idx >> 6, d = idx & 63;
    s_k[t * KSTRIDE + d] = (_Float16)k[(size_t)b * T_ * D_ + idx];
  }
  for (int idx = tid; idx < 8 * KSTRIDE; idx += 256)   // zero pad rows 200..207
    s_k[200 * KSTRIDE + idx] = (_Float16)0.f;

  for (int idx = tid; idx < 256 * H1_; idx += 256) {   // W1^T
    int f = idx / H1_, h = idx % H1_;
    s_w1t[h * W1STRIDE + f] = (_Float16)W1[idx];
  }
  for (int idx = tid; idx < 48 * 96; idx += 256) {     // W2^T zero-padded
    int h2 = idx / 96, h1 = idx % 96;
    _Float16 val = (_Float16)0.f;
    if (h2 < H2_ && h1 < H1_) val = (_Float16)W2[h1 * H2_ + h2];
    s_w2t[h2 * W2STRIDE + h1] = val;
  }
  for (int idx = tid; idx < TPAD * HSTRIDE; idx += 256)
    s_h[idx] = (_Float16)0.f;

  if (tid < H1_) s_b1[tid] = b1[tid];
  if (tid < 48) {
    s_b2[tid] = (tid < H2_) ? b2[tid] : 0.f;
    s_wf[tid] = (tid < H2_) ? Wf[tid] : 0.f;
  }
  if (tid == 0) s_bf = bf[0];
  __syncthreads();

  // ---------------- Per-wave dual-tile pipeline ----------------
  const int lrow  = lane & 15;          // row within tile (A) or column (B/C)
  const int khalf = (lane >> 4) * 8;    // A-layout K sub-offset: 0 or 8
  const int koffb = (lane >> 4) * 16;   // unused; B loads use khalf like A
  const int mofs  = (lane >> 4) * 8;    // C-layout row offset: 0 or 8

  // Hoist per-lane constants (column-indexed) out of the hot loops.
  float b1c[NT1], b2c[NT2], wfc[NT2];
#pragma unroll
  for (int n = 0; n < NT1; ++n) b1c[n] = s_b1[n * 16 + lrow];
#pragma unroll
  for (int n = 0; n < NT2; ++n) { b2c[n] = s_b2[n * 16 + lrow]; wfc[n] = s_wf[n * 16 + lrow]; }
  const float bfv = s_bf;

  // Wave wv owns tiles wv and wv+8 (tile1 exists only for wv<5); both tiles share
  // every B-fragment load, halving LDS traffic per WMMA.
  const bool has1 = (wv + 8) < 13;
  const int  rb0  = wv * 16;
  const int  rb1  = has1 ? (wv + 8) * 16 : rb0;  // clamp: compute duplicate, discard
  const int  tr0  = rb0 + lrow;
  const int  tr1  = rb1 + lrow;

  // k/q fragments (A layout) for the two 32-wide halves of D
  v16h kf0[2], kf1[2], qf[2];
#pragma unroll
  for (int h = 0; h < 2; ++h) {
    const int dbase = h * 32;
    v8h k0lo = *(const v8h*)&s_k[tr0 * KSTRIDE + dbase + khalf];
    v8h k0hi = *(const v8h*)&s_k[tr0 * KSTRIDE + dbase + khalf + 16];
    v8h k1lo = *(const v8h*)&s_k[tr1 * KSTRIDE + dbase + khalf];
    v8h k1hi = *(const v8h*)&s_k[tr1 * KSTRIDE + dbase + khalf + 16];
    v8h qlo  = *(const v8h*)&s_q[dbase + khalf];
    v8h qhi  = *(const v8h*)&s_q[dbase + khalf + 16];
#pragma unroll
    for (int i = 0; i < 8; ++i) {
      kf0[h][i] = k0lo[i]; kf0[h][i + 8] = k0hi[i];
      kf1[h][i] = k1lo[i]; kf1[h][i + 8] = k1hi[i];
      qf[h][i]  = qlo[i];  qf[h][i + 8]  = qhi[i];
    }
  }

  // -------- GEMM1: X[16,256] x W1[256,80], two row-tiles at once --------
  v8f acc0[NT1] = {}, acc1[NT1] = {};
#pragma unroll
  for (int s = 0; s < NK1; ++s) {
    const int h = s & 1, r = s >> 1;
    v16h a0, a1;
    if      (r == 0) { a0 = qf[h];           a1 = qf[h];           }
    else if (r == 1) { a0 = kf0[h];          a1 = kf1[h];          }
    else if (r == 2) { a0 = qf[h] - kf0[h];  a1 = qf[h] - kf1[h];  }
    else             { a0 = qf[h] * kf0[h];  a1 = qf[h] * kf1[h];  }
    const int kcol = 32 * s + khalf;
#pragma unroll
    for (int n = 0; n < NT1; ++n) {
      const int nr = n * 16 + lrow;
      v8h blo = *(const v8h*)&s_w1t[nr * W1STRIDE + kcol];
      v8h bhi = *(const v8h*)&s_w1t[nr * W1STRIDE + kcol + 16];
      v16h bb;
#pragma unroll
      for (int i = 0; i < 8; ++i) { bb[i] = blo[i]; bb[i + 8] = bhi[i]; }
      acc0[n] = wmma_f16(a0, bb, acc0[n]);
      acc1[n] = wmma_f16(a1, bb, acc1[n]);
    }
  }

  // bias + relu -> H (f16, row-major, zero-padded K to 96)
#pragma unroll
  for (int n = 0; n < NT1; ++n) {
    const int col = n * 16 + lrow;
#pragma unroll
    for (int g = 0; g < 8; ++g) {
      float v0 = acc0[n][g] + b1c[n];
      v0 = v0 > 0.f ? v0 : 0.f;
      s_h[(rb0 + mofs + g) * HSTRIDE + col] = (_Float16)v0;
      if (has1) {
        float v1 = acc1[n][g] + b1c[n];
        v1 = v1 > 0.f ? v1 : 0.f;
        s_h[(rb1 + mofs + g) * HSTRIDE + col] = (_Float16)v1;
      }
    }
  }

  // -------- GEMM2: H[16,96] x W2[96,48], two row-tiles at once --------
  v8f c0[NT2] = {}, c1[NT2] = {};
#pragma unroll
  for (int s = 0; s < NK2; ++s) {
    v8h a0lo = *(const v8h*)&s_h[tr0 * HSTRIDE + 32 * s + khalf];
    v8h a0hi = *(const v8h*)&s_h[tr0 * HSTRIDE + 32 * s + khalf + 16];
    v8h a1lo = *(const v8h*)&s_h[tr1 * HSTRIDE + 32 * s + khalf];
    v8h a1hi = *(const v8h*)&s_h[tr1 * HSTRIDE + 32 * s + khalf + 16];
    v16h a0, a1;
#pragma unroll
    for (int i = 0; i < 8; ++i) {
      a0[i] = a0lo[i]; a0[i + 8] = a0hi[i];
      a1[i] = a1lo[i]; a1[i + 8] = a1hi[i];
    }
    const int kcol = 32 * s + khalf;
#pragma unroll
    for (int n = 0; n < NT2; ++n) {
      const int nr = n * 16 + lrow;
      v8h blo = *(const v8h*)&s_w2t[nr * W2STRIDE + kcol];
      v8h bhi = *(const v8h*)&s_w2t[nr * W2STRIDE + kcol + 16];
      v16h bb;
#pragma unroll
      for (int i = 0; i < 8; ++i) { bb[i] = blo[i]; bb[i + 8] = bhi[i]; }
      c0[n] = wmma_f16(a0, bb, c0[n]);
      c1[n] = wmma_f16(a1, bb, c1[n]);
    }
  }

  // -------- logits: relu(G+b2).Wf + bf via in-register shfl reduction --------
  // Row M's 48 columns live across the 16 lanes of one half-wave at VGPR g=M%8;
  // xor-shuffles with masks 1,2,4,8 reduce entirely within the half.
#pragma unroll
  for (int g = 0; g < 8; ++g) {
    float r0 = 0.f, r1 = 0.f;
#pragma unroll
    for (int n = 0; n < NT2; ++n) {
      float u0 = c0[n][g] + b2c[n];
      u0 = u0 > 0.f ? u0 : 0.f;
      r0 += u0 * wfc[n];
      float u1 = c1[n][g] + b2c[n];
      u1 = u1 > 0.f ? u1 : 0.f;
      r1 += u1 * wfc[n];
    }
#pragma unroll
    for (int m = 1; m <= 8; m <<= 1) {
      r0 += __shfl_xor(r0, m, 32);
      r1 += __shfl_xor(r1, m, 32);
    }
    if (lrow == 0) {
      const int t0 = rb0 + mofs + g;
      if (t0 < T_) s_logits[t0] = r0 + bfv;
      if (has1) {
        const int t1 = rb1 + mofs + g;
        if (t1 < T_) s_logits[t1] = r1 + bfv;
      }
    }
  }
  __syncthreads();

  // ---------------- Masked softmax over T ----------------
  const bool active = (tid < T_);
  float lg = 0.f, val = -3.4e38f;
  if (active) {
    lg = s_logits[tid];
    if (mask[(size_t)b * T_ + tid] == 0) lg = -4294967295.0f;
    val = lg;
  }
  s_red[tid] = val;
  __syncthreads();
  for (int off = 128; off > 0; off >>= 1) {
    if (tid < off) { float o = s_red[tid + off]; if (o > s_red[tid]) s_red[tid] = o; }
    __syncthreads();
  }
  const float m = s_red[0];
  __syncthreads();
  const float e = active ? __expf(lg - m) : 0.f;
  s_red[tid] = e;
  __syncthreads();
  for (int off = 128; off > 0; off >>= 1) {
    if (tid < off) s_red[tid] += s_red[tid + off];
    __syncthreads();
  }
  const float inv = 1.f / s_red[0];
  __syncthreads();
  if (active) s_logits[tid] = e * inv;
  __syncthreads();

  // ---------------- out[b] = attn . v[b] ----------------
  const int grp = tid >> 6;     // 0..3 : T split 4x50
  const int d   = tid & 63;
  float accv = 0.f;
  for (int t = grp * 50; t < grp * 50 + 50; ++t)
    accv += s_logits[t] * vb[t * D_ + d];
  s_vpart[grp][d] = accv;
  __syncthreads();
  if (tid < 64)
    out[(size_t)b * D_ + tid] =
        s_vpart[0][tid] + s_vpart[1][tid] + s_vpart[2][tid] + s_vpart[3][tid];
}

extern "C" void kernel_launch(void* const* d_in, const int* in_sizes, int n_in,
                              void* d_out, int out_size, void* d_ws, size_t ws_size,
                              hipStream_t stream) {
  const float* q    = (const float*)d_in[0];
  const float* k    = (const float*)d_in[1];
  const float* v    = (const float*)d_in[2];
  const int*   mask = (const int*)  d_in[3];
  const float* W1   = (const float*)d_in[4];
  const float* b1   = (const float*)d_in[5];
  const float* W2   = (const float*)d_in[6];
  const float* b2   = (const float*)d_in[7];
  const float* Wf   = (const float*)d_in[8];
  const float* bf   = (const float*)d_in[9];
  float* out = (float*)d_out;

  attn_mlp_kernel<<<B_, 256, 0, stream>>>(q, k, v, mask, W1, b1, W2, b2, Wf, bf, out);
}
